// Attention_49005576847767
// MI455X (gfx1250) — matrix operand order repaired and checked by reading, not compiled, not run on hardware
//
#include <hip/hip_runtime.h>
#include <hip/hip_bf16.h>

// ---------------- types ----------------
typedef __bf16 bf16;
typedef __attribute__((ext_vector_type(16))) __bf16 v16bf;
typedef __attribute__((ext_vector_type(8)))  __bf16 v8bf;
typedef __attribute__((ext_vector_type(8)))  float  v8f;

#define S_LEN   2048
#define H_DIM   4096
#define HQ_N    32
#define HKV_N   8
#define D_HEAD  128
#define WINDOW  1024

static __device__ inline v16bf cat8(v8bf lo, v8bf hi) {
  return __builtin_shufflevector(lo, hi, 0,1,2,3,4,5,6,7,8,9,10,11,12,13,14,15);
}

// Low 32 bits of a flat pointer into __shared__ == wave-relative LDS byte address
// (ISA 10.2: LDS aperture keeps the LDS offset in addr[31:0]).
static __device__ inline unsigned lds_u32(const void* p) {
  return (unsigned)(unsigned long long)p;
}

// CDNA5 async global->LDS copy (no VGPR round trip, tracked by ASYNCcnt).
// INST_OFFSET is added to BOTH the global and LDS addresses (ISA 10.7.4).
template <int OFS>
static __device__ inline void async_cp_b128(unsigned lds, const void* g) {
  asm volatile("global_load_async_to_lds_b128 %0, %1, off offset:%2"
               :: "v"(lds), "v"(g), "n"(OFS) : "memory");
}
static __device__ inline void async_wait0() {
  asm volatile("s_wait_asynccnt 0x0" ::: "memory");
}

// ---------------- fp32 -> bf16 convert ----------------
__global__ __launch_bounds__(256) void cvt_bf16_kernel(const float* __restrict__ in,
                                                       bf16* __restrict__ out, int n) {
  int i = (blockIdx.x * 256 + threadIdx.x) * 4;
  if (i >= n) return;
  float4 v = *(const float4*)(in + i);
  out[i + 0] = (bf16)v.x; out[i + 1] = (bf16)v.y;
  out[i + 2] = (bf16)v.z; out[i + 3] = (bf16)v.w;
}

// ---------------- transpose + convert: W[K][N] fp32 -> WT[N][K] bf16 ----------------
__global__ __launch_bounds__(256) void transpose_cvt_kernel(const float* __restrict__ W,
                                                            bf16* __restrict__ WT,
                                                            int K, int N) {
  __shared__ float tile[32][33];
  int n0 = blockIdx.x * 32, k0 = blockIdx.y * 32;
  int r = threadIdx.x >> 5, c = threadIdx.x & 31;
#pragma unroll
  for (int i = 0; i < 4; ++i)
    tile[c][r + 8 * i] = W[(size_t)(k0 + r + 8 * i) * N + n0 + c];
  __syncthreads();
#pragma unroll
  for (int i = 0; i < 4; ++i)
    WT[(size_t)(n0 + r + 8 * i) * K + k0 + c] = (bf16)tile[r + 8 * i][c];
}

// ---------------- tiled WMMA GEMM: C[M][N] = A[M][K] * BT[N][K]^T ----------------
// block = 256 threads = 8 waves (4x2), tile 128x128, K-step 32,
// double-buffered LDS fed by global_load_async_to_lds_b128.
__global__ __launch_bounds__(256) void gemm_bf16_kernel(const bf16* __restrict__ A,
                                                        const bf16* __restrict__ BT,
                                                        float* __restrict__ C,
                                                        int M, int N, int K) {
  __shared__ __align__(32) bf16 As[2][128][32];
  __shared__ __align__(32) bf16 Bs[2][128][32];
  int tid = threadIdx.x;
  int lane = tid & 31, wave = tid >> 5;
  int wm = wave >> 1, wn = wave & 1;
  int lnN = lane & 15, kg = lane >> 4;
  int m0 = blockIdx.y * 128, n0 = blockIdx.x * 128;

  v8f acc[2][4];
#pragma unroll
  for (int a = 0; a < 2; ++a)
#pragma unroll
    for (int b = 0; b < 4; ++b) acc[a][b] = {};

  int sr = tid >> 1;           // 0..127 staging row
  int sc = (tid & 1) * 16;     // staging K offset (this thread copies 32B of A, 32B of B)
  const bf16* Ag = A  + (size_t)(m0 + sr) * K + sc;
  const bf16* Bg = BT + (size_t)(n0 + sr) * K + sc;
  unsigned ldsA[2] = { lds_u32(&As[0][sr][sc]), lds_u32(&As[1][sr][sc]) };
  unsigned ldsB[2] = { lds_u32(&Bs[0][sr][sc]), lds_u32(&Bs[1][sr][sc]) };

  auto stage = [&](int buf, int k0) {
    async_cp_b128<0>(ldsA[buf], Ag + k0);
    async_cp_b128<16>(ldsA[buf], Ag + k0);
    async_cp_b128<0>(ldsB[buf], Bg + k0);
    async_cp_b128<16>(ldsB[buf], Bg + k0);
  };

  stage(0, 0);
  async_wait0();
  __syncthreads();

  int buf = 0;
  for (int k0 = 0; k0 < K; k0 += 32) {
    if (k0 + 32 < K) {
      stage(buf ^ 1, k0 + 32);                 // overlap copy of next tile
      __builtin_prefetch(Ag + k0 + 64, 0, 1);  // warm L2 two tiles ahead
      __builtin_prefetch(Bg + k0 + 64, 0, 1);
    }

    v16bf af[2], bfg[4];
#pragma unroll
    for (int fr = 0; fr < 2; ++fr) {
      int r = wm * 32 + fr * 16 + lnN;
      af[fr] = cat8(*(const v8bf*)&As[buf][r][kg * 8],
                    *(const v8bf*)&As[buf][r][16 + kg * 8]);
    }
#pragma unroll
    for (int fc = 0; fc < 4; ++fc) {
      int c = wn * 64 + fc * 16 + lnN;
      bfg[fc] = cat8(*(const v8bf*)&Bs[buf][c][kg * 8],
                     *(const v8bf*)&Bs[buf][c][16 + kg * 8]);
    }
#pragma unroll
    for (int fr = 0; fr < 2; ++fr)
#pragma unroll
      for (int fc = 0; fc < 4; ++fc)
        acc[fr][fc] = __builtin_amdgcn_wmma_f32_16x16x32_bf16(
            false, af[fr], false, bfg[fc], (short)0, acc[fr][fc], false, false);

    async_wait0();      // next tile fully in LDS
    __syncthreads();
    buf ^= 1;
  }

#pragma unroll
  for (int fr = 0; fr < 2; ++fr)
#pragma unroll
    for (int fc = 0; fc < 4; ++fc) {
      int col = n0 + wn * 64 + fc * 16 + lnN;
#pragma unroll
      for (int j = 0; j < 8; ++j) {
        int row = m0 + wm * 32 + fr * 16 + kg * 8 + j;
        C[(size_t)row * N + col] = acc[fr][fc][j];
      }
    }
}

// ---------------- RMSNorm + RoPE for Q: qf[S][HQ*D] -> qb[HQ][S][D] bf16 ----------------
__global__ __launch_bounds__(256) void q_norm_rope_kernel(const float* __restrict__ qf,
                                                          const float* __restrict__ w,
                                                          bf16* __restrict__ qb) {
  int row = blockIdx.x * 8 + (threadIdx.x >> 5);
  int lane = threadIdx.x & 31;
  int s = row >> 5;     // / HQ
  int h = row & 31;
  int d0 = lane * 4;
  float4 x  = *(const float4*)(qf + (size_t)s * (HQ_N * D_HEAD) + h * D_HEAD + d0);
  float4 ww = *(const float4*)(w + d0);
  float ss = x.x * x.x + x.y * x.y + x.z * x.z + x.w * x.w;
#pragma unroll
  for (int off = 16; off >= 1; off >>= 1) ss += __shfl_xor(ss, off, 32);
  float r = rsqrtf(ss * (1.0f / 128.0f) + 1e-6f);
  float n[4] = { x.x * r * ww.x, x.y * r * ww.y, x.z * r * ww.z, x.w * r * ww.w };
  const float kNegLT = -0.14391156f; // -ln(10000)/64
  bf16* dst = qb + ((size_t)h * S_LEN + s) * D_HEAD + d0;
#pragma unroll
  for (int j = 0; j < 4; ++j) {
    float p = __shfl_xor(n[j], 16, 32);
    float f = (float)((lane & 15) * 4 + j);
    float ang = (float)s * __expf(f * kNegLT);
    float cs = __cosf(ang), sn = __sinf(ang);
    float o = (lane < 16) ? (n[j] * cs - p * sn) : (n[j] * cs + p * sn);
    dst[j] = (bf16)o;
  }
}

// ---------------- RMSNorm+RoPE for K and transpose-convert V ----------------
__global__ __launch_bounds__(256) void kv_norm_rope_kernel(const float* __restrict__ kf,
                                                           const float* __restrict__ vf,
                                                           const float* __restrict__ w,
                                                           bf16* __restrict__ kbuf,
                                                           bf16* __restrict__ vtb) {
  int row = blockIdx.x * 8 + (threadIdx.x >> 5);
  int lane = threadIdx.x & 31;
  int s = row >> 3;     // / HKV
  int h = row & 7;
  int d0 = lane * 4;
  float4 x  = *(const float4*)(kf + (size_t)s * (HKV_N * D_HEAD) + h * D_HEAD + d0);
  float4 ww = *(const float4*)(w + d0);
  float ss = x.x * x.x + x.y * x.y + x.z * x.z + x.w * x.w;
#pragma unroll
  for (int off = 16; off >= 1; off >>= 1) ss += __shfl_xor(ss, off, 32);
  float r = rsqrtf(ss * (1.0f / 128.0f) + 1e-6f);
  float n[4] = { x.x * r * ww.x, x.y * r * ww.y, x.z * r * ww.z, x.w * r * ww.w };
  const float kNegLT = -0.14391156f;
  bf16* dst = kbuf + ((size_t)h * S_LEN + s) * D_HEAD + d0;
#pragma unroll
  for (int j = 0; j < 4; ++j) {
    float p = __shfl_xor(n[j], 16, 32);
    float f = (float)((lane & 15) * 4 + j);
    float ang = (float)s * __expf(f * kNegLT);
    float cs = __cosf(ang), sn = __sinf(ang);
    float o = (lane < 16) ? (n[j] * cs - p * sn) : (n[j] * cs + p * sn);
    dst[j] = (bf16)o;
  }
  float4 v = *(const float4*)(vf + (size_t)s * (HKV_N * D_HEAD) + h * D_HEAD + d0);
  vtb[((size_t)h * D_HEAD + d0 + 0) * S_LEN + s] = (bf16)v.x;
  vtb[((size_t)h * D_HEAD + d0 + 1) * S_LEN + s] = (bf16)v.y;
  vtb[((size_t)h * D_HEAD + d0 + 2) * S_LEN + s] = (bf16)v.z;
  vtb[((size_t)h * D_HEAD + d0 + 3) * S_LEN + s] = (bf16)v.w;
}

// ---------------- flash attention: sliding-window causal, tanh softcap, GQA ----------------
// grid (S/128, HQ); block 256 = 8 waves; each wave owns 16 q rows.
// K/V^T tiles double-buffered in LDS via async copies.
__global__ __launch_bounds__(256) void attn_kernel(const bf16* __restrict__ qb,
                                                   const bf16* __restrict__ kbuf,
                                                   const bf16* __restrict__ vtb,
                                                   bf16* __restrict__ ob) {
  __shared__ __align__(32) bf16 Ks[2][64][128];   // keys x D
  __shared__ __align__(32) bf16 Vt[2][128][64];   // D x keys
  __shared__ __align__(32) bf16 Ps[8][16][64];    // per-wave P scratch
  int tid = threadIdx.x, lane = tid & 31, wave = tid >> 5;
  int lnN = lane & 15, kg = lane >> 4;
  int qs = blockIdx.x * 128;
  int hq = blockIdx.y;
  int kv = hq >> 2;                 // G = 4
  int qrow0 = qs + wave * 16;

  // Q fragments stay in registers (A-layout, 4 K-chunks of 32 cover D=128)
  v16bf qfrag[4];
  {
    const bf16* qp = qb + ((size_t)hq * S_LEN + qrow0 + lnN) * D_HEAD;
#pragma unroll
    for (int kk = 0; kk < 4; ++kk)
      qfrag[kk] = cat8(*(const v8bf*)(qp + kk * 32 + kg * 8),
                       *(const v8bf*)(qp + kk * 32 + 16 + kg * 8));
  }

  v8f out[8];
#pragma unroll
  for (int fd = 0; fd < 8; ++fd) out[fd] = {};
  float m_i[8], l_i[8];
#pragma unroll
  for (int j = 0; j < 8; ++j) { m_i[j] = -1e30f; l_i[j] = 0.f; }

  int t0 = (qs >= WINDOW) ? ((qs - WINDOW) >> 6) : 0;
  int t1 = (qs + 127) >> 6;
  const float scale = 0.088388347648318447f;   // 1/sqrt(128)

  // staging geometry: thread copies 64B of K and 64B of V^T per tile
  int kr = tid >> 2, kc = (tid & 3) * 32;
  int vr = tid >> 1, vc = (tid & 1) * 32;
  unsigned ldsK[2] = { lds_u32(&Ks[0][kr][kc]), lds_u32(&Ks[1][kr][kc]) };
  unsigned ldsV[2] = { lds_u32(&Vt[0][vr][vc]), lds_u32(&Vt[1][vr][vc]) };

  auto stage = [&](int bufi, int kb0) {
    const bf16* ksrc = kbuf + ((size_t)kv * S_LEN + kb0 + kr) * D_HEAD + kc;
    const bf16* vsrc = vtb + ((size_t)kv * D_HEAD + vr) * S_LEN + kb0 + vc;
    async_cp_b128<0>(ldsK[bufi], ksrc);
    async_cp_b128<16>(ldsK[bufi], ksrc);
    async_cp_b128<32>(ldsK[bufi], ksrc);
    async_cp_b128<48>(ldsK[bufi], ksrc);
    async_cp_b128<0>(ldsV[bufi], vsrc);
    async_cp_b128<16>(ldsV[bufi], vsrc);
    async_cp_b128<32>(ldsV[bufi], vsrc);
    async_cp_b128<48>(ldsV[bufi], vsrc);
  };

  stage(0, t0 << 6);
  async_wait0();
  __syncthreads();

  int buf = 0;
  for (int kt = t0; kt <= t1; ++kt) {
    int kb0 = kt << 6;
    if (kt < t1) stage(buf ^ 1, kb0 + 64);   // overlap next K/V tile copy

    // scores S = Q K^T (16 rows x 64 keys per wave)
    v8f sc[4];
#pragma unroll
    for (int fc = 0; fc < 4; ++fc) {
      v8f a = {};
#pragma unroll
      for (int kk = 0; kk < 4; ++kk) {
        v16bf bfrag = cat8(*(const v8bf*)&Ks[buf][fc * 16 + lnN][kk * 32 + kg * 8],
                           *(const v8bf*)&Ks[buf][fc * 16 + lnN][kk * 32 + 16 + kg * 8]);
        a = __builtin_amdgcn_wmma_f32_16x16x32_bf16(false, qfrag[kk], false, bfrag,
                                                    (short)0, a, false, false);
      }
      sc[fc] = a;
    }
    // softcap + window/causal mask
#pragma unroll
    for (int fc = 0; fc < 4; ++fc) {
      int jk = kb0 + fc * 16 + lnN;
#pragma unroll
      for (int j = 0; j < 8; ++j) {
        int i = qrow0 + kg * 8 + j;
        float v = sc[fc][j] * scale;
        v = 50.0f * tanhf(v * (1.0f / 50.0f));
        bool valid = (jk <= i) && (i - jk <= WINDOW);
        sc[fc][j] = valid ? v : -1e30f;
      }
    }
    // online softmax per row (rows live across the 16-lane halves)
    float alpha[8];
#pragma unroll
    for (int j = 0; j < 8; ++j) {
      float mx = fmaxf(fmaxf(sc[0][j], sc[1][j]), fmaxf(sc[2][j], sc[3][j]));
#pragma unroll
      for (int off = 8; off >= 1; off >>= 1) mx = fmaxf(mx, __shfl_xor(mx, off, 32));
      float mnew = fmaxf(m_i[j], mx);
      alpha[j] = __expf(m_i[j] - mnew);
      m_i[j] = mnew;
      float rs = 0.f;
#pragma unroll
      for (int fc = 0; fc < 4; ++fc) {
        float p = (sc[fc][j] > -1e29f) ? __expf(sc[fc][j] - mnew) : 0.f;
        sc[fc][j] = p;
        rs += p;
      }
#pragma unroll
      for (int off = 8; off >= 1; off >>= 1) rs += __shfl_xor(rs, off, 32);
      l_i[j] = l_i[j] * alpha[j] + rs;
    }
#pragma unroll
    for (int fd = 0; fd < 8; ++fd)
#pragma unroll
      for (int j = 0; j < 8; ++j) out[fd][j] *= alpha[j];

    // P (C layout) -> LDS -> A layout for P*V
#pragma unroll
    for (int fc = 0; fc < 4; ++fc)
#pragma unroll
      for (int j = 0; j < 8; ++j)
        Ps[wave][kg * 8 + j][fc * 16 + lnN] = (bf16)sc[fc][j];
    __syncthreads();

#pragma unroll
    for (int kk = 0; kk < 2; ++kk) {
      v16bf pf = cat8(*(const v8bf*)&Ps[wave][lnN][kk * 32 + kg * 8],
                      *(const v8bf*)&Ps[wave][lnN][kk * 32 + 16 + kg * 8]);
#pragma unroll
      for (int fd = 0; fd < 8; ++fd) {
        v16bf vfrag = cat8(*(const v8bf*)&Vt[buf][fd * 16 + lnN][kk * 32 + kg * 8],
                           *(const v8bf*)&Vt[buf][fd * 16 + lnN][kk * 32 + 16 + kg * 8]);
        out[fd] = __builtin_amdgcn_wmma_f32_16x16x32_bf16(false, pf, false, vfrag,
                                                          (short)0, out[fd], false, false);
      }
    }
    async_wait0();      // next K/V tile resident
    __syncthreads();
    buf ^= 1;
  }

  // finalize: divide by l, store bf16 into ob[S][HQ*D]
#pragma unroll
  for (int fd = 0; fd < 8; ++fd)
#pragma unroll
    for (int j = 0; j < 8; ++j) {
      int row = qrow0 + kg * 8 + j;
      float v = out[fd][j] / l_i[j];
      ob[(size_t)row * (HQ_N * D_HEAD) + hq * D_HEAD + fd * 16 + lnN] = (bf16)v;
    }
}

// ---------------- host launch ----------------
extern "C" void kernel_launch(void* const* d_in, const int* in_sizes, int n_in,
                              void* d_out, int out_size, void* d_ws, size_t ws_size,
                              hipStream_t stream) {
  (void)in_sizes; (void)n_in; (void)out_size; (void)ws_size;
  const float* x   = (const float*)d_in[0];
  const float* wq  = (const float*)d_in[1];
  const float* wk  = (const float*)d_in[2];
  const float* wv  = (const float*)d_in[3];
  const float* wo  = (const float*)d_in[4];
  const float* qnw = (const float*)d_in[5];
  const float* knw = (const float*)d_in[6];
  float* out = (float*)d_out;

  // workspace carve-up (256B aligned)
  char* ws = (char*)d_ws;
  size_t off = 0;
  auto take = [&](size_t bytes) -> char* {
    off = (off + 255) & ~(size_t)255;
    char* p = ws + off;
    off += bytes;
    return p;
  };
  const size_t SH  = (size_t)S_LEN * H_DIM;           // 8 Mi elems
  const size_t SKV = (size_t)S_LEN * HKV_N * D_HEAD;  // 2 Mi elems
  bf16*  xb   = (bf16*)take(SH * 2);
  bf16*  wqT  = (bf16*)take((size_t)H_DIM * H_DIM * 2);
  bf16*  wkT  = (bf16*)take((size_t)H_DIM * (HKV_N * D_HEAD) * 2);
  bf16*  wvT  = (bf16*)take((size_t)H_DIM * (HKV_N * D_HEAD) * 2);
  bf16*  woT  = (bf16*)take((size_t)H_DIM * H_DIM * 2);
  float* qf   = (float*)take(SH * 4);
  float* kf   = (float*)take(SKV * 4);
  float* vf   = (float*)take(SKV * 4);
  bf16*  qb   = (bf16*)take(SH * 2);
  bf16*  kbuf = (bf16*)take(SKV * 2);
  bf16*  vtb  = (bf16*)take(SKV * 2);
  bf16*  ob   = (bf16*)take(SH * 2);

  // 1. convert x to bf16
  cvt_bf16_kernel<<<(int)(SH / 1024), 256, 0, stream>>>(x, xb, (int)SH);
  // 2. transpose+convert weights: W[K][N] -> WT[N][K] bf16
  transpose_cvt_kernel<<<dim3(H_DIM / 32, H_DIM / 32), 256, 0, stream>>>(wq, wqT, H_DIM, H_DIM);
  transpose_cvt_kernel<<<dim3((HKV_N * D_HEAD) / 32, H_DIM / 32), 256, 0, stream>>>(wk, wkT, H_DIM, HKV_N * D_HEAD);
  transpose_cvt_kernel<<<dim3((HKV_N * D_HEAD) / 32, H_DIM / 32), 256, 0, stream>>>(wv, wvT, H_DIM, HKV_N * D_HEAD);
  transpose_cvt_kernel<<<dim3(H_DIM / 32, H_DIM / 32), 256, 0, stream>>>(wo, woT, H_DIM, H_DIM);
  // 3. QKV projections (WMMA)
  gemm_bf16_kernel<<<dim3(H_DIM / 128, S_LEN / 128), 256, 0, stream>>>(xb, wqT, qf, S_LEN, H_DIM, H_DIM);
  gemm_bf16_kernel<<<dim3((HKV_N * D_HEAD) / 128, S_LEN / 128), 256, 0, stream>>>(xb, wkT, kf, S_LEN, HKV_N * D_HEAD, H_DIM);
  gemm_bf16_kernel<<<dim3((HKV_N * D_HEAD) / 128, S_LEN / 128), 256, 0, stream>>>(xb, wvT, vf, S_LEN, HKV_N * D_HEAD, H_DIM);
  // 4. RMSNorm + RoPE, layout shuffles to head-major bf16
  q_norm_rope_kernel<<<(S_LEN * HQ_N) / 8, 256, 0, stream>>>(qf, qnw, qb);
  kv_norm_rope_kernel<<<(S_LEN * HKV_N) / 8, 256, 0, stream>>>(kf, vf, knw, kbuf, vtb);
  // 5. flash attention (WMMA for QK^T and PV)
  attn_kernel<<<dim3(S_LEN / 128, HQ_N), 256, 0, stream>>>(qb, kbuf, vtb, ob);
  // 6. output projection (WMMA) -> fp32 result
  gemm_bf16_kernel<<<dim3(H_DIM / 128, S_LEN / 128), 256, 0, stream>>>(ob, woT, out, S_LEN, H_DIM, H_DIM);
}
